// MultiHeadAttention_10471130268305
// MI455X (gfx1250) — compile-verified, matrix operand order checked by reading, not run on hardware
//
#include <hip/hip_runtime.h>

// ---------------------------------------------------------------------------
// MultiHeadAttention for MI455X (gfx1250): bf16 WMMA pipeline, double-buffered
// B=4, S=1024, E=1024, H=16, D=64
// ---------------------------------------------------------------------------

#define B_   4
#define S_   1024
#define E_   1024
#define H_   16
#define D_   64
#define LDA  40   // LDS stride (halves) for 32-wide K tiles; 80B -> conflict-free b128 reads
#define LDV  72   // LDS stride (halves) for 64-wide tiles;  144B -> conflict-free b128 reads

typedef __attribute__((ext_vector_type(16))) __bf16 bf16x16;
typedef __attribute__((ext_vector_type(4)))  __bf16 bf16x4;
typedef __attribute__((ext_vector_type(8)))  float  v8f;
typedef __attribute__((ext_vector_type(4)))  float  f32x4;

union Frag {
    bf16x16 bf;
    f32x4   f4[2];
};

// ---- CDNA5 async global->LDS copy (ASYNCcnt path), guarded ----------------
#if __has_builtin(__builtin_amdgcn_global_load_async_to_lds_b128) && \
    __has_builtin(__builtin_amdgcn_s_wait_asynccnt)
#define USE_ASYNC 1
#define AS1 __attribute__((address_space(1)))
#define AS3 __attribute__((address_space(3)))
typedef int i32x4v __attribute__((vector_size(16)));   // matches builtin param type
__device__ __forceinline__ void cp_async16(void* lds, const void* g) {
    __builtin_amdgcn_global_load_async_to_lds_b128(
        (AS1 i32x4v*)g, (AS3 i32x4v*)lds, 0, 0);
}
#else
#define USE_ASYNC 0
#endif

// ---------------------------------------------------------------------------
// GEMM: X(fp32, M x K) @ W(fp32, K x N) + bias -> bf16 Out.
// transOut=0 -> (B,H,S,D) layout; transOut=1 -> (B,H,D,S) layout (for V).
// tile 64x64, 128 threads (4 waves), K-step 32, double-buffered LDS.
// ---------------------------------------------------------------------------
__global__ __launch_bounds__(128)
void mha_gemm_qkv(const float* __restrict__ X, const float* __restrict__ W,
                  const float* __restrict__ bias, __bf16* __restrict__ Out,
                  int transOut)
{
    __shared__ __align__(16) __bf16 As[2][64 * LDA];
    __shared__ __align__(16) __bf16 Bs[2][64 * LDA];   // transposed: [n][k]

    const int tid  = threadIdx.x;
    const int lane = tid & 31;
    const int wave = tid >> 5;       // M sub-tile (0..3)
    const int l16  = lane & 15;
    const int hl   = lane >> 4;
    const int kb   = hl * 8;
    const int n0   = blockIdx.x * 64;
    const int m0   = blockIdx.y * 64;

    v8f zero = {};
    v8f acc[4];
#pragma unroll
    for (int f = 0; f < 4; ++f) acc[f] = zero;

    f32x4 ra[4], rb[4];
    auto loadTiles = [&](int k0) {
#pragma unroll
        for (int i = 0; i < 4; ++i) {
            int lin = tid + i * 128;
            int r = lin >> 3, c = (lin & 7) << 2;
            ra[i] = *(const f32x4*)(X + (size_t)(m0 + r) * E_ + k0 + c);
        }
#pragma unroll
        for (int i = 0; i < 4; ++i) {
            int lin = tid + i * 128;
            int kk = lin >> 4, c = (lin & 15) << 2;
            rb[i] = *(const f32x4*)(W + (size_t)(k0 + kk) * E_ + n0 + c);
        }
    };
    auto storeTiles = [&](int buf) {
#pragma unroll
        for (int i = 0; i < 4; ++i) {
            int lin = tid + i * 128;
            int r = lin >> 3, c = (lin & 7) << 2;
            bf16x4 p = { (__bf16)ra[i].x, (__bf16)ra[i].y,
                         (__bf16)ra[i].z, (__bf16)ra[i].w };
            *(bf16x4*)(&As[buf][r * LDA + c]) = p;     // one ds_store_b64
        }
#pragma unroll
        for (int i = 0; i < 4; ++i) {
            int lin = tid + i * 128;
            int kk = lin >> 4, c = (lin & 15) << 2;
            Bs[buf][(c + 0) * LDA + kk] = (__bf16)rb[i].x;
            Bs[buf][(c + 1) * LDA + kk] = (__bf16)rb[i].y;
            Bs[buf][(c + 2) * LDA + kk] = (__bf16)rb[i].z;
            Bs[buf][(c + 3) * LDA + kk] = (__bf16)rb[i].w;
        }
    };

    loadTiles(0);
    storeTiles(0);
    __syncthreads();

    for (int k0 = 0, it = 0; k0 < E_; k0 += 32, ++it) {
        const int  buf  = it & 1;
        const bool more = (k0 + 32) < E_;
        if (more) loadTiles(k0 + 32);               // overlap with WMMAs below

        Frag a;
        const __bf16* ap = &As[buf][(wave * 16 + l16) * LDA + kb];
        a.f4[0] = *(const f32x4*)(ap);
        a.f4[1] = *(const f32x4*)(ap + 16);
#pragma unroll
        for (int f = 0; f < 4; ++f) {
            Frag b;
            const __bf16* bp = &Bs[buf][(f * 16 + l16) * LDA + kb];
            b.f4[0] = *(const f32x4*)(bp);
            b.f4[1] = *(const f32x4*)(bp + 16);
            acc[f] = __builtin_amdgcn_wmma_f32_16x16x32_bf16(
                false, a.bf, false, b.bf, (short)0, acc[f], false, false);
        }

        if (more) storeTiles(buf ^ 1);
        __syncthreads();
    }

    // epilogue: + bias, scatter to (B,H,S,D) or (B,H,D,S)
#pragma unroll
    for (int f = 0; f < 4; ++f) {
        int gn = n0 + f * 16 + l16;
        float bs = bias[gn];
        int h = gn >> 6, d = gn & 63;
#pragma unroll
        for (int r = 0; r < 8; ++r) {
            int gm = m0 + wave * 16 + r + 8 * hl;
            int b = gm >> 10, s = gm & (S_ - 1);
            __bf16 val = (__bf16)(acc[f][r] + bs);
            if (transOut)
                Out[(((size_t)b * H_ + h) * D_ + d) * S_ + s] = val;
            else
                Out[(((size_t)b * H_ + h) * S_ + s) * D_ + d] = val;
        }
    }
}

// ---------------------------------------------------------------------------
// GEMM: X(bf16, M x K) @ W(fp32, K x N) + bias -> fp32 Out (row-major M x N)
// ---------------------------------------------------------------------------
__global__ __launch_bounds__(128)
void mha_gemm_out(const __bf16* __restrict__ X, const float* __restrict__ W,
                  const float* __restrict__ bias, float* __restrict__ Out)
{
    __shared__ __align__(16) __bf16 As[2][64 * LDA];
    __shared__ __align__(16) __bf16 Bs[2][64 * LDA];

    const int tid  = threadIdx.x;
    const int lane = tid & 31;
    const int wave = tid >> 5;
    const int l16  = lane & 15;
    const int hl   = lane >> 4;
    const int kb   = hl * 8;
    const int n0   = blockIdx.x * 64;
    const int m0   = blockIdx.y * 64;

    v8f zero = {};
    v8f acc[4];
#pragma unroll
    for (int f = 0; f < 4; ++f) acc[f] = zero;

    f32x4 ra[2], rb[4];
    auto loadTiles = [&](int k0) {
#pragma unroll
        for (int i = 0; i < 2; ++i) {
            int lin = tid + i * 128;           // 8-half chunks (256 total)
            int r = lin >> 2, c = (lin & 3) << 3;
            ra[i] = *(const f32x4*)(X + (size_t)(m0 + r) * E_ + k0 + c);
        }
#pragma unroll
        for (int i = 0; i < 4; ++i) {
            int lin = tid + i * 128;
            int kk = lin >> 4, c = (lin & 15) << 2;
            rb[i] = *(const f32x4*)(W + (size_t)(k0 + kk) * E_ + n0 + c);
        }
    };
    auto storeTiles = [&](int buf) {
#pragma unroll
        for (int i = 0; i < 2; ++i) {
            int lin = tid + i * 128;
            int r = lin >> 2, c = (lin & 3) << 3;
            *(f32x4*)(&As[buf][r * LDA + c]) = ra[i];
        }
#pragma unroll
        for (int i = 0; i < 4; ++i) {
            int lin = tid + i * 128;
            int kk = lin >> 4, c = (lin & 15) << 2;
            Bs[buf][(c + 0) * LDA + kk] = (__bf16)rb[i].x;
            Bs[buf][(c + 1) * LDA + kk] = (__bf16)rb[i].y;
            Bs[buf][(c + 2) * LDA + kk] = (__bf16)rb[i].z;
            Bs[buf][(c + 3) * LDA + kk] = (__bf16)rb[i].w;
        }
    };

    loadTiles(0);
    storeTiles(0);
    __syncthreads();

    for (int k0 = 0, it = 0; k0 < E_; k0 += 32, ++it) {
        const int  buf  = it & 1;
        const bool more = (k0 + 32) < E_;
        if (more) loadTiles(k0 + 32);

        Frag a;
        const __bf16* ap = &As[buf][(wave * 16 + l16) * LDA + kb];
        a.f4[0] = *(const f32x4*)(ap);
        a.f4[1] = *(const f32x4*)(ap + 16);
#pragma unroll
        for (int f = 0; f < 4; ++f) {
            Frag b;
            const __bf16* bp = &Bs[buf][(f * 16 + l16) * LDA + kb];
            b.f4[0] = *(const f32x4*)(bp);
            b.f4[1] = *(const f32x4*)(bp + 16);
            acc[f] = __builtin_amdgcn_wmma_f32_16x16x32_bf16(
                false, a.bf, false, b.bf, (short)0, acc[f], false, false);
        }

        if (more) storeTiles(buf ^ 1);
        __syncthreads();
    }

#pragma unroll
    for (int f = 0; f < 4; ++f) {
        int gn = n0 + f * 16 + l16;
        float bs = bias[gn];
#pragma unroll
        for (int r = 0; r < 8; ++r) {
            int gm = m0 + wave * 16 + r + 8 * hl;
            Out[(size_t)gm * E_ + gn] = acc[f][r] + bs;
        }
    }
}

// ---------------------------------------------------------------------------
// RoPE in place on bf16 tensor in (B,H,S,D) layout. One thread per (d, d+32).
// ---------------------------------------------------------------------------
__global__ void mha_rope(__bf16* __restrict__ T, const float* __restrict__ cosT,
                         const float* __restrict__ sinT, int total)
{
    int idx = blockIdx.x * blockDim.x + threadIdx.x;
    if (idx >= total) return;
    int dd = idx & 31;
    int s  = (idx >> 5) & (S_ - 1);
    size_t base = (size_t)(idx >> 5) * D_;
    float x1 = (float)T[base + dd];
    float x2 = (float)T[base + dd + 32];
    float c1 = cosT[s * D_ + dd],      s1 = sinT[s * D_ + dd];
    float c2 = cosT[s * D_ + dd + 32], s2 = sinT[s * D_ + dd + 32];
    T[base + dd]      = (__bf16)(x1 * c1 - x2 * s1);
    T[base + dd + 32] = (__bf16)(x2 * c2 + x1 * s2);
}

// ---------------------------------------------------------------------------
// Flash attention: one block per (b, h, 64-row query tile). 4 waves.
// K in (B,H,S,D); V pre-transposed in (B,H,D,S) -> both tiles stage into LDS
// as contiguous 16B chunks (async global->LDS when available), double-buffered.
// ---------------------------------------------------------------------------
__global__ __launch_bounds__(128)
void mha_attention(const __bf16* __restrict__ Q, const __bf16* __restrict__ Kg,
                   const __bf16* __restrict__ Vg, __bf16* __restrict__ Ob)
{
    __shared__ __align__(16) __bf16 Ks[2][64 * LDV];  // K tile [j][d]
    __shared__ __align__(16) __bf16 Vs[2][64 * LDV];  // V tile [d][j]
    __shared__ __align__(16) __bf16 Ps[64 * LDV];     // P tile [row][j]

    const int tid  = threadIdx.x;
    const int lane = tid & 31;
    const int wave = tid >> 5;
    const int l16  = lane & 15;
    const int hl   = lane >> 4;
    const int kb   = hl * 8;

    const int bid = blockIdx.x;
    const int qt  = bid & (S_ / 64 - 1);
    const int bh  = bid >> 4;                  // b*H + h
    const int h   = bh & (H_ - 1);
    const int b   = bh >> 4;

    const __bf16* Qh = Q  + (size_t)bh * S_ * D_;
    const __bf16* Kh = Kg + (size_t)bh * S_ * D_;
    const __bf16* Vh = Vg + (size_t)bh * D_ * S_;    // transposed layout

    // stage K tile [j][d] and V tile [d][j] for key-tile jt into buffer `buf`
    auto stage = [&](int buf, int jt) {
        const int j0 = jt * 64;
#pragma unroll
        for (int i = 0; i < 4; ++i) {
            int lin = tid + i * 128;           // 512 chunks of 16B per tile
            int r = lin >> 3, c8 = (lin & 7) << 3;
#if USE_ASYNC
            cp_async16(&Ks[buf][r * LDV + c8], Kh + (size_t)(j0 + r) * D_ + c8);
            cp_async16(&Vs[buf][r * LDV + c8], Vh + (size_t)r * S_ + j0 + c8);
#else
            *(f32x4*)(&Ks[buf][r * LDV + c8]) =
                *(const f32x4*)(Kh + (size_t)(j0 + r) * D_ + c8);
            *(f32x4*)(&Vs[buf][r * LDV + c8]) =
                *(const f32x4*)(Vh + (size_t)r * S_ + j0 + c8);
#endif
        }
    };

    // loop-invariant Q fragments (wave's 16 rows, two 32-wide d chunks)
    Frag qa[2];
    {
        const __bf16* qp = Qh + (size_t)(qt * 64 + wave * 16 + l16) * D_;
#pragma unroll
        for (int c = 0; c < 2; ++c) {
            qa[c].f4[0] = *(const f32x4*)(qp + c * 32 + kb);
            qa[c].f4[1] = *(const f32x4*)(qp + c * 32 + kb + 16);
        }
    }

    float mrow[8], lrow[8];
    v8f zero = {};
    v8f o[4];
#pragma unroll
    for (int r = 0; r < 8; ++r) { mrow[r] = -3.0e38f; lrow[r] = 0.f; }
#pragma unroll
    for (int f = 0; f < 4; ++f) o[f] = zero;

    stage(0, 0);                                  // prologue

    for (int jt = 0; jt <= qt; ++jt) {
        const int  buf  = jt & 1;
        const bool more = jt < qt;
        if (more) stage(buf ^ 1, jt + 1);         // prefetch next K/V tiles
#if USE_ASYNC
        if (more) __builtin_amdgcn_s_wait_asynccnt(8);
        else      __builtin_amdgcn_s_wait_asynccnt(0);
#endif
        __syncthreads();                          // current tiles visible

        // S = Q @ K^T
        v8f sc[4];
#pragma unroll
        for (int f = 0; f < 4; ++f) sc[f] = zero;
#pragma unroll
        for (int f = 0; f < 4; ++f) {
            const __bf16* kp = &Ks[buf][(f * 16 + l16) * LDV];
#pragma unroll
            for (int c = 0; c < 2; ++c) {
                Frag kf;
                kf.f4[0] = *(const f32x4*)(kp + c * 32 + kb);
                kf.f4[1] = *(const f32x4*)(kp + c * 32 + kb + 16);
                sc[f] = __builtin_amdgcn_wmma_f32_16x16x32_bf16(
                    false, qa[c].bf, false, kf.bf, (short)0, sc[f], false, false);
            }
        }

        const float scale = 0.125f;               // 1/sqrt(64)
        const bool diag = (jt == qt);
#pragma unroll
        for (int f = 0; f < 4; ++f)
#pragma unroll
            for (int r = 0; r < 8; ++r) {
                float v = sc[f][r] * scale;
                int qr = wave * 16 + r + 8 * hl;
                int jc = f * 16 + l16;
                if (diag && (jc > qr)) v = -3.0e38f;   // causal (select, no branch)
                sc[f][r] = v;
            }

        // online softmax: row stats over 16-lane halves
        float mnew[8], alpha[8], rsum[8];
#pragma unroll
        for (int r = 0; r < 8; ++r) {
            float v = fmaxf(fmaxf(sc[0][r], sc[1][r]), fmaxf(sc[2][r], sc[3][r]));
#pragma unroll
            for (int m = 1; m < 16; m <<= 1) v = fmaxf(v, __shfl_xor(v, m, 32));
            mnew[r] = fmaxf(mrow[r], v);
            alpha[r] = __expf(mrow[r] - mnew[r]);
            mrow[r] = mnew[r];
            rsum[r] = 0.f;
        }
#pragma unroll
        for (int f = 0; f < 4; ++f)
#pragma unroll
            for (int r = 0; r < 8; ++r) {
                float p = __expf(sc[f][r] - mnew[r]);
                sc[f][r] = p;
                rsum[r] += p;
                Ps[(wave * 16 + r + 8 * hl) * LDV + f * 16 + l16] = (__bf16)p;
            }
#pragma unroll
        for (int r = 0; r < 8; ++r) {
            float v = rsum[r];
#pragma unroll
            for (int m = 1; m < 16; m <<= 1) v += __shfl_xor(v, m, 32);
            lrow[r] = lrow[r] * alpha[r] + v;
        }
#pragma unroll
        for (int f = 0; f < 4; ++f)
#pragma unroll
            for (int r = 0; r < 8; ++r) o[f][r] *= alpha[r];

        // O += P @ V (P re-read from LDS in A layout; own-wave rows only)
#pragma unroll
        for (int c = 0; c < 2; ++c) {
            Frag pa;
            const __bf16* pp = Ps + (wave * 16 + l16) * LDV + c * 32 + kb;
            pa.f4[0] = *(const f32x4*)(pp);
            pa.f4[1] = *(const f32x4*)(pp + 16);
#pragma unroll
            for (int f = 0; f < 4; ++f) {
                Frag vb;
                const __bf16* vp = &Vs[buf][(f * 16 + l16) * LDV + c * 32 + kb];
                vb.f4[0] = *(const f32x4*)(vp);
                vb.f4[1] = *(const f32x4*)(vp + 16);
                o[f] = __builtin_amdgcn_wmma_f32_16x16x32_bf16(
                    false, pa.bf, false, vb.bf, (short)0, o[f], false, false);
            }
        }
        __syncthreads();                          // done reading buf before reuse
    }

    // epilogue: normalize, store to (B,S,E) bf16
#pragma unroll
    for (int f = 0; f < 4; ++f) {
        int d = f * 16 + l16;
#pragma unroll
        for (int r = 0; r < 8; ++r) {
            int qr = qt * 64 + wave * 16 + r + 8 * hl;
            float inv = 1.f / lrow[r];
            Ob[((size_t)b * S_ + qr) * E_ + h * D_ + d] = (__bf16)(o[f][r] * inv);
        }
    }
}

// ---------------------------------------------------------------------------
// Host launcher
// ---------------------------------------------------------------------------
extern "C" void kernel_launch(void* const* d_in, const int* in_sizes, int n_in,
                              void* d_out, int out_size, void* d_ws, size_t ws_size,
                              hipStream_t stream)
{
    const float* query = (const float*)d_in[0];
    const float* key   = (const float*)d_in[1];
    const float* value = (const float*)d_in[2];
    const float* cosT  = (const float*)d_in[3];
    const float* sinT  = (const float*)d_in[4];
    // d_in[5] = attn_mask (causal) — implemented analytically in-kernel
    const float* Wq = (const float*)d_in[6];
    const float* bq = (const float*)d_in[7];
    const float* Wk = (const float*)d_in[8];
    const float* bk = (const float*)d_in[9];
    const float* Wv = (const float*)d_in[10];
    const float* bv = (const float*)d_in[11];
    const float* Wo = (const float*)d_in[12];
    const float* bo = (const float*)d_in[13];
    float* out = (float*)d_out;

    const size_t nQKV = (size_t)B_ * H_ * S_ * D_;   // 4M elements
    __bf16* Qb = (__bf16*)d_ws;
    __bf16* Kb = Qb + nQKV;
    __bf16* Vb = Kb + nQKV;      // (B,H,D,S) transposed
    __bf16* Ob = Vb + nQKV;      // (B,S,E)

    dim3 gGrid(E_ / 64, (B_ * S_) / 64);             // 16 x 64
    dim3 gBlk(128);

    mha_gemm_qkv<<<gGrid, gBlk, 0, stream>>>(query, Wq, bq, Qb, 0);
    mha_gemm_qkv<<<gGrid, gBlk, 0, stream>>>(key,   Wk, bk, Kb, 0);
    mha_gemm_qkv<<<gGrid, gBlk, 0, stream>>>(value, Wv, bv, Vb, 1);

    int ropeTotal = B_ * H_ * S_ * (D_ / 2);         // 2M
    mha_rope<<<(ropeTotal + 255) / 256, 256, 0, stream>>>(Qb, cosT, sinT, ropeTotal);
    mha_rope<<<(ropeTotal + 255) / 256, 256, 0, stream>>>(Kb, cosT, sinT, ropeTotal);

    mha_attention<<<B_ * H_ * (S_ / 64), gBlk, 0, stream>>>(Qb, Kb, Vb, Ob);

    mha_gemm_out<<<gGrid, gBlk, 0, stream>>>(Ob, Wo, bo, out);
}
